// CQTGenerator_41137196761190
// MI455X (gfx1250) — compile-verified, matrix operand order checked by reading, not run on hardware
//
#include <hip/hip_runtime.h>
#include <cmath>

#define SRATE   16000
#define HOP     512
#define NBINS   84
#define NGROUPS 11           // ceil(84/8) groups of 8 bins sharing one K axis
#define LWAVE   2560000
#define NFRAMES 5000
#define LDS_HALFS 21504      // max window: 15*512 + roundup32(wl0) = 21152 halfs

typedef _Float16 h16;
typedef __attribute__((ext_vector_type(8)))  h16   v8h;
typedef __attribute__((ext_vector_type(16))) h16   v16h;
typedef __attribute__((ext_vector_type(8)))  float v8f;

struct CqtParams {
    int wl[NBINS];
    int fb[NBINS];
    int pad[NBINS];
    int jmin[NGROUPS];        // K-axis origin (absolute sample offset), 8-aligned
    int kchunks[NGROUPS];     // number of K=32 chunks for this group
    long long boff[NGROUPS];  // half-element offset of this group's B tiles in ws
    int totalChunks;
};

__device__ __forceinline__ int reflect_idx(int j) {
    j = (j < 0) ? -j : j;                      // torch reflect: x[-1] -> x[1]
    j = (j >= LWAVE) ? (2 * LWAVE - 2 - j) : j;
    return j;
}

// ---- kernel 1a: waveform f32 -> f16 staging copy -------------------------
__global__ void cqt_wave_to_h(const float* __restrict__ x, h16* __restrict__ xh) {
    int i = blockIdx.x * blockDim.x + threadIdx.x;
    int stride = gridDim.x * blockDim.x;
    for (; i < LWAVE; i += stride) xh[i] = (h16)x[i];
}

// ---- kernel 1b: build B tiles (windowed DFT-bin kernels) in WMMA-B layout -
// Layout per K=32 chunk: [lane 0..31][elem 0..15] halfs.
// lanes 0-15 : column N=lane,    elems -> K {0..7, 16..23}
// lanes 16-31: column N=lane-16, elems -> K {8..15, 24..31}
// Column N = 2*b + (0=re / 1=im) for bin (8*g + b).
__global__ void cqt_fill_b(h16* __restrict__ ws, CqtParams p) {
    long long tid = (long long)blockIdx.x * blockDim.x + threadIdx.x;
    long long total = (long long)p.totalChunks * 512;
    if (tid >= total) return;
    int cid = (int)(tid >> 9);          // global chunk id
    int r   = (int)(tid & 511);         // element within chunk
    int g = 0, acc = 0;
    for (int gg = 0; gg < NGROUPS; ++gg) {
        if (cid < acc + p.kchunks[gg]) { g = gg; break; }
        acc += p.kchunks[gg];
    }
    int kc   = cid - acc;
    int lane = r >> 4, e = r & 15;
    int hl   = lane & 15, hi = lane >> 4;
    int N    = hl;
    int K    = (e < 8 ? e : e + 8) + hi * 8;
    int j    = p.jmin[g] + kc * 32 + K; // absolute sample offset on shared K axis
    int b    = N >> 1;
    int k    = g * 8 + b;
    float val = 0.f;
    if (k < NBINS) {
        int wl = p.wl[k];
        int n  = j + p.pad[k];
        if (wl >= 32 && n >= 0 && n < wl) {
            const float TWOPI = 6.28318530717958647692f;
            float w = 0.5f - 0.5f * cosf(TWOPI * (float)n / (float)wl);
            // exact integer phase reduction keeps cos/sin accurate at large n
            long long m = ((long long)p.fb[k] * (long long)n) % (long long)wl;
            float ang = TWOPI * (float)m / (float)wl;
            val = (N & 1) ? (-w * sinf(ang)) : (w * cosf(ang));
        }
    }
    ws[p.boff[g] + (long long)kc * 512 + r] = (h16)val;
}

// ---- kernel 2: 16-frame x 16-column WMMA GEMM per (frame tile, bin group) -
__global__ void __launch_bounds__(32)
cqt_wmma(const h16* __restrict__ xh, const h16* __restrict__ ws,
         float* __restrict__ out, CqtParams p) {
    __shared__ __align__(16) h16 lds[LDS_HALFS];
    const int lane = threadIdx.x;            // one wave32 per workgroup
    const int g    = blockIdx.y;
    const int t0   = blockIdx.x * 16;
    const int jmin = p.jmin[g];
    const int kch  = p.kchunks[g];
    const int winLen = 15 * HOP + kch * 32;  // contiguous A window (multiple of 32)
    const int winStart = t0 * HOP + jmin;

    // Stage the shared window into LDS: aligned b128 loads in the interior,
    // per-element reflect only at the signal edges.
    const int nvec = winLen >> 3;
    for (int c = lane; c < nvec; c += 32) {
        int j0 = winStart + (c << 3);        // jmin is 8-aligned -> 16B aligned
        v8h v;
        if (j0 >= 0 && j0 + 7 < LWAVE) {
            v = *(const v8h*)(xh + j0);
        } else {
            for (int i = 0; i < 8; ++i) v[i] = xh[reflect_idx(j0 + i)];
        }
        *(v8h*)(&lds[c << 3]) = v;
    }
    __syncthreads();

    const h16* bptr = ws + p.boff[g];
    const int hl = lane & 15, hi = lane >> 4;
    v8f acc = {};
    for (int kc = 0; kc < kch; ++kc) {
        // A fragment: row M = hl, K subset {0..7,16..23} (hi=0) / {8..15,24..31} (hi=1)
        int base = hl * HOP + kc * 32 + hi * 8;   // 16B aligned in halfs*2
        v8h a0 = *(const v8h*)(&lds[base]);
        v8h a1 = *(const v8h*)(&lds[base + 16]);
        v16h A;
        for (int i = 0; i < 8; ++i) { A[i] = a0[i]; A[i + 8] = a1[i]; }
        // B fragment: 32 contiguous bytes per lane (precomputed layout)
        v16h B = *(const v16h*)(bptr + (long long)kc * 512 + lane * 16);
        acc = __builtin_amdgcn_wmma_f32_16x16x32_f16(
            false, A, false, B, (short)0, acc, false, false);
    }

    // Epilogue: pair (re, im) columns via wave32 shuffle, write magnitudes.
    for (int v = 0; v < 8; ++v) {
        float mine  = acc[v];
        float other = __shfl_xor(mine, 1, 32);
        if ((hl & 1) == 0) {
            int bin   = g * 8 + (hl >> 1);
            int frame = t0 + v + hi * 8;       // D layout: M = v + 8*hi
            if (bin < NBINS && frame < NFRAMES)
                out[bin * NFRAMES + frame] = sqrtf(mine * mine + other * other);
        }
    }
}

extern "C" void kernel_launch(void* const* d_in, const int* in_sizes, int n_in,
                              void* d_out, int out_size, void* d_ws, size_t ws_size,
                              hipStream_t stream) {
    (void)in_sizes; (void)n_in; (void)out_size; (void)ws_size;
    const float* wave = (const float*)d_in[0];
    float* out = (float*)d_out;
    h16* ws = (h16*)d_ws;

    CqtParams p;
    const double Q = 1.0 / (std::pow(2.0, 1.0 / 12.0) - 1.0);
    for (int k = 0; k < NBINS; ++k) {
        double freq = 20.0 * std::pow(2.0, (double)k / 12.0);
        int wl = (int)((double)SRATE * Q / freq);
        if (wl > LWAVE / 4) wl = LWAVE / 4;
        int fb = (int)(freq * (double)wl / (double)SRATE);
        if (fb > wl / 2) fb = wl / 2;
        p.wl[k] = wl; p.fb[k] = fb; p.pad[k] = wl / 2;
    }
    int chunkAcc = 0;
    for (int g = 0; g < NGROUPS; ++g) {
        int wl0  = p.wl[g * 8];
        int pad0 = p.pad[g * 8];
        int jmin = -(((pad0 + 7) / 8) * 8);          // 8-aligned for b128 loads
        int kspanRaw = (wl0 - pad0) - jmin;          // covers every bin in group
        int kch = (kspanRaw + 31) / 32;
        int maxK = (LDS_HALFS - 15 * HOP) / 32;      // LDS capacity clamp
        if (kch > maxK) kch = maxK;
        p.jmin[g] = jmin;
        p.kchunks[g] = kch;
        p.boff[g] = (long long)LWAVE + (long long)chunkAcc * 512;
        chunkAcc += kch;
    }
    p.totalChunks = chunkAcc;

    hipLaunchKernelGGL(cqt_wave_to_h, dim3(2048), dim3(256), 0, stream, wave, ws);
    long long totalB = (long long)chunkAcc * 512;
    int gridB = (int)((totalB + 255) / 256);
    hipLaunchKernelGGL(cqt_fill_b, dim3(gridB), dim3(256), 0, stream, ws, p);
    dim3 grid((NFRAMES + 15) / 16, NGROUPS);
    hipLaunchKernelGGL(cqt_wmma, grid, dim3(32), 0, stream,
                       (const h16*)ws, (const h16*)ws, out, p);
}
